// Seq2Seq_36026185679097
// MI455X (gfx1250) — matrix operand order repaired and checked by reading, not compiled, not run on hardware
//
#include <hip/hip_runtime.h>
#include <hip/hip_bf16.h>

#define H 256
#define H3 768
#define T 2048
#define VOUT 32000

typedef __attribute__((ext_vector_type(16))) __bf16 v16bf;
typedef __attribute__((ext_vector_type(8)))  float  v8f;

union BFrag {
    v16bf v;
    unsigned int d[8];
};

__device__ __forceinline__ unsigned short f2bf(float f) {
    unsigned int u = __float_as_uint(f);
    unsigned int r = (u + 0x7FFFu + ((u >> 16) & 1u)) >> 16;   // RNE
    return (unsigned short)r;
}
// packed-bf16 -> f32 expansion (bf16 is just the top 16 bits of f32)
__device__ __forceinline__ float bflo(unsigned int u) { return __uint_as_float(u << 16); }
__device__ __forceinline__ float bfhi(unsigned int u) { return __uint_as_float(u & 0xFFFF0000u); }

__device__ __forceinline__ float sigm(float x) {
    return 1.0f / (1.0f + expf(-x));
}

// ---------------------------------------------------------------- converts
__global__ void k_f32_to_bf16(const float* __restrict__ src,
                              unsigned short* __restrict__ dst, int n) {
    int i = blockIdx.x * blockDim.x + threadIdx.x;
    if (i < n) dst[i] = f2bf(src[i]);
}

// gather embedding rows per token, optional relu, emit bf16 (T x 256)
__global__ void k_gather_bf16(const float* __restrict__ emb,
                              const int* __restrict__ tok,
                              unsigned short* __restrict__ dst, int do_relu) {
    int t = blockIdx.x;
    int i = threadIdx.x;                       // 256 threads
    float v = emb[tok[t] * H + i];
    if (do_relu) v = fmaxf(v, 0.0f);
    dst[t * H + i] = f2bf(v);
}

// ---------------------------------------------------------------- WMMA GEMM
// C(M x N) = A(M x 256) * W(N x 256)^T + bias, A/W bf16 row-major, C fp32.
// Register-blocked: each wave computes a 32(M) x 64(N) macro-tile
//   (2 A-frags x 4 B-frags -> 8 WMMAs per K-step of 32).
// blockDim = 128 (4 waves -> 4 consecutive 64-wide N slabs = 256 N per block).
__global__ void __launch_bounds__(128)
k_wmma_gemm_bf16(const unsigned short* __restrict__ A,
                 const unsigned short* __restrict__ W,
                 const float* __restrict__ bias,
                 float* __restrict__ C, int N) {
    const int lane = threadIdx.x & 31;
    const int wave = threadIdx.x >> 5;
    const int j0 = (blockIdx.x * 4 + wave) * 64;   // N slab origin (64 wide)
    const int i0 = blockIdx.y * 32;                // M slab origin (32 tall)

    const int mr = lane & 15;       // tile row (A) / tile col (B)
    const int kh = lane >> 4;       // K-half select

    // fold bias into accumulator init: C[m][j0+jt*16+n] starts at bias[col]
    v8f acc[2][4];
#pragma unroll
    for (int jt = 0; jt < 4; ++jt) {
        float bv = bias[j0 + jt * 16 + mr];
#pragma unroll
        for (int v = 0; v < 8; ++v) { acc[0][jt][v] = bv; acc[1][jt][v] = bv; }
    }

    const unsigned short* arow0 = A + (i0 + mr) * H;        // M tile 0
    const unsigned short* arow1 = arow0 + 16 * H;           // M tile 1
    const unsigned short* wrow[4];
#pragma unroll
    for (int jt = 0; jt < 4; ++jt)
        wrow[jt] = W + (j0 + jt * 16 + mr) * H;

#pragma unroll
    for (int kb = 0; kb < H; kb += 32) {
        BFrag a0, a1, b[4];
#pragma unroll
        for (int v = 0; v < 4; ++v) {
            // A 16x32 bf16 layout: VGPR v<4  -> K = kb + kh*8 + 2v
            //                      VGPR v>=4 -> K = kb + 16 + kh*8 + 2(v-4)
            a0.d[v]     = *(const unsigned int*)(arow0 + kb + kh * 8 + 2 * v);
            a0.d[v + 4] = *(const unsigned int*)(arow0 + kb + 16 + kh * 8 + 2 * v);
            a1.d[v]     = *(const unsigned int*)(arow1 + kb + kh * 8 + 2 * v);
            a1.d[v + 4] = *(const unsigned int*)(arow1 + kb + 16 + kh * 8 + 2 * v);
        }
#pragma unroll
        for (int jt = 0; jt < 4; ++jt)
#pragma unroll
            for (int v = 0; v < 8; ++v) {
                // B 32x16 bf16 layout: same K order as A (v<4: kh*8+2v, v>=4: 16+kh*8+2(v-4))
                b[jt].d[v] = *(const unsigned int*)(wrow[jt] + kb + (v >> 2) * 16 + kh * 8 + 2 * (v & 3));
            }
#pragma unroll
        for (int jt = 0; jt < 4; ++jt) {
            acc[0][jt] = __builtin_amdgcn_wmma_f32_16x16x32_bf16(
                false, a0.v, false, b[jt].v, (short)0, acc[0][jt], false, false);
            acc[1][jt] = __builtin_amdgcn_wmma_f32_16x16x32_bf16(
                false, a1.v, false, b[jt].v, (short)0, acc[1][jt], false, false);
        }
    }

    // C/D 16x16 f32 layout: n = lane&15, VGPR v -> M = v + 8*(lane>=16)
    const int n = lane & 15;
    const int mb = kh * 8;
#pragma unroll
    for (int mi = 0; mi < 2; ++mi)
#pragma unroll
        for (int jt = 0; jt < 4; ++jt)
#pragma unroll
            for (int v = 0; v < 8; ++v)
                C[(i0 + mi * 16 + mb + v) * N + j0 + jt * 16 + n] = acc[mi][jt][v];
}

// ---------------------------------------------------------------- GRU scan
// Single persistent workgroup (768 threads = 24 wave32). h lives in LDS,
// W_hh streamed as bf16 from L2 each step (384 KB/half-step, L2-resident),
// expanded to f32 in-register; all accumulation and gate math in fp32.
__global__ void __launch_bounds__(768)
k_gru_seq(const float* __restrict__ gi_enc, const float* __restrict__ gi_dec,
          const unsigned short* __restrict__ Wenc, const float* __restrict__ benc,
          const unsigned short* __restrict__ Wdec, const float* __restrict__ bdec,
          unsigned short* __restrict__ Hbf) {
    __shared__ float h[H];
    __shared__ float gh[H3];
    const int tid = threadIdx.x;           // 0..767

    if (tid < H) h[tid] = 0.0f;
    __syncthreads();

    const uint4* wre = (const uint4*)(Wenc + tid * H);   // 32 x uint4 = 256 bf16
    const uint4* wrd = (const uint4*)(Wdec + tid * H);
    const float be = benc[tid];
    const float bd = bdec[tid];

    for (int t = 0; t < T; ++t) {
        __builtin_prefetch(gi_enc + (t + 1) * H3, 0, 3);
        __builtin_prefetch(gi_dec + (t + 1) * H3, 0, 3);

        // ---- encoder half-step:  gh = W_hh @ h + b_hh
        float acc = be;
#pragma unroll 8
        for (int k = 0; k < H / 8; ++k) {
            uint4 w = wre[k];
            const float* hv = h + k * 8;
            acc += bflo(w.x) * hv[0] + bfhi(w.x) * hv[1]
                 + bflo(w.y) * hv[2] + bfhi(w.y) * hv[3]
                 + bflo(w.z) * hv[4] + bfhi(w.z) * hv[5]
                 + bflo(w.w) * hv[6] + bfhi(w.w) * hv[7];
        }
        gh[tid] = acc;
        __syncthreads();
        if (tid < H) {
            const float* gi = gi_enc + t * H3;
            float r = sigm(gi[tid] + gh[tid]);
            float z = sigm(gi[H + tid] + gh[H + tid]);
            float nn = tanhf(gi[2 * H + tid] + r * gh[2 * H + tid]);
            h[tid] = (1.0f - z) * nn + z * h[tid];
        }
        __syncthreads();

        // ---- decoder half-step
        acc = bd;
#pragma unroll 8
        for (int k = 0; k < H / 8; ++k) {
            uint4 w = wrd[k];
            const float* hv = h + k * 8;
            acc += bflo(w.x) * hv[0] + bfhi(w.x) * hv[1]
                 + bflo(w.y) * hv[2] + bfhi(w.y) * hv[3]
                 + bflo(w.z) * hv[4] + bfhi(w.z) * hv[5]
                 + bflo(w.w) * hv[6] + bfhi(w.w) * hv[7];
        }
        gh[tid] = acc;
        __syncthreads();
        if (tid < H) {
            const float* gi = gi_dec + t * H3;
            float r = sigm(gi[tid] + gh[tid]);
            float z = sigm(gi[H + tid] + gh[H + tid]);
            float nn = tanhf(gi[2 * H + tid] + r * gh[2 * H + tid]);
            float hn = (1.0f - z) * nn + z * h[tid];
            h[tid] = hn;
            Hbf[t * H + tid] = f2bf(hn);   // bf16 copy feeds logits GEMM
        }
        __syncthreads();
    }
}

// ---------------------------------------------------------------- softmax
__global__ void __launch_bounds__(256)
k_logsoftmax(float* __restrict__ out) {
    __shared__ float red[256];
    float* p = out + (long)blockIdx.x * VOUT;
    const int tid = threadIdx.x;

    float m = -INFINITY;
    for (int j = tid; j < VOUT; j += 256) m = fmaxf(m, p[j]);
    red[tid] = m;
    __syncthreads();
    for (int s = 128; s > 0; s >>= 1) {
        if (tid < s) red[tid] = fmaxf(red[tid], red[tid + s]);
        __syncthreads();
    }
    m = red[0];
    __syncthreads();

    float sum = 0.0f;
    for (int j = tid; j < VOUT; j += 256) sum += expf(p[j] - m);
    red[tid] = sum;
    __syncthreads();
    for (int s = 128; s > 0; s >>= 1) {
        if (tid < s) red[tid] += red[tid + s];
        __syncthreads();
    }
    const float lse = m + logf(red[0]);

    for (int j = tid; j < VOUT; j += 256) p[j] -= lse;
}

// ---------------------------------------------------------------- launcher
extern "C" void kernel_launch(void* const* d_in, const int* in_sizes, int n_in,
                              void* d_out, int out_size, void* d_ws, size_t ws_size,
                              hipStream_t stream) {
    const int*   src      = (const int*)d_in[0];
    const int*   trg      = (const int*)d_in[1];
    const float* enc_emb  = (const float*)d_in[2];
    const float* enc_W_ih = (const float*)d_in[3];
    const float* enc_W_hh = (const float*)d_in[4];
    const float* enc_b_ih = (const float*)d_in[5];
    const float* enc_b_hh = (const float*)d_in[6];
    const float* dec_emb  = (const float*)d_in[7];
    const float* dec_W_ih = (const float*)d_in[8];
    const float* dec_W_hh = (const float*)d_in[9];
    const float* dec_b_ih = (const float*)d_in[10];
    const float* dec_b_hh = (const float*)d_in[11];
    const float* out_W    = (const float*)d_in[12];
    const float* out_b    = (const float*)d_in[13];
    float* out = (float*)d_out;

    // workspace carve-up (all sizes multiples of 256 bytes)
    char* ws = (char*)d_ws;
    unsigned short* Wout_bf = (unsigned short*)ws;             ws += (size_t)VOUT * H * 2;  // 16.38 MB
    unsigned short* Xenc_bf = (unsigned short*)ws;             ws += (size_t)T * H * 2;     // 1 MB
    unsigned short* Xdec_bf = (unsigned short*)ws;             ws += (size_t)T * H * 2;
    unsigned short* WihE_bf = (unsigned short*)ws;             ws += (size_t)H3 * H * 2;
    unsigned short* WihD_bf = (unsigned short*)ws;             ws += (size_t)H3 * H * 2;
    unsigned short* WhhE_bf = (unsigned short*)ws;             ws += (size_t)H3 * H * 2;    // 384 KB
    unsigned short* WhhD_bf = (unsigned short*)ws;             ws += (size_t)H3 * H * 2;
    float*          gi_enc  = (float*)ws;                      ws += (size_t)T * H3 * 4;    // 6.29 MB
    float*          gi_dec  = (float*)ws;                      ws += (size_t)T * H3 * 4;
    unsigned short* Hbf     = (unsigned short*)ws;             ws += (size_t)T * H * 2;

    // 1) precision conversion + embedding gathers (fully parallel)
    k_f32_to_bf16<<<(VOUT * H + 255) / 256, 256, 0, stream>>>(out_W, Wout_bf, VOUT * H);
    k_f32_to_bf16<<<(H3 * H + 255) / 256, 256, 0, stream>>>(enc_W_ih, WihE_bf, H3 * H);
    k_f32_to_bf16<<<(H3 * H + 255) / 256, 256, 0, stream>>>(dec_W_ih, WihD_bf, H3 * H);
    k_f32_to_bf16<<<(H3 * H + 255) / 256, 256, 0, stream>>>(enc_W_hh, WhhE_bf, H3 * H);
    k_f32_to_bf16<<<(H3 * H + 255) / 256, 256, 0, stream>>>(dec_W_hh, WhhD_bf, H3 * H);
    k_gather_bf16<<<T, H, 0, stream>>>(enc_emb, src, Xenc_bf, 0);
    k_gather_bf16<<<T, H, 0, stream>>>(dec_emb, trg, Xdec_bf, 1);

    // 2) batched input-gate GEMMs: gi = X @ W_ih^T + b_ih   (2048 x 768)
    {
        dim3 grid(H3 / 256, T / 32);               // 3 x 64
        k_wmma_gemm_bf16<<<grid, 128, 0, stream>>>(Xenc_bf, WihE_bf, enc_b_ih, gi_enc, H3);
        k_wmma_gemm_bf16<<<grid, 128, 0, stream>>>(Xdec_bf, WihD_bf, dec_b_ih, gi_dec, H3);
    }

    // 3) sequential recurrence (latency-bound, one persistent workgroup)
    k_gru_seq<<<1, H3, 0, stream>>>(gi_enc, gi_dec, WhhE_bf, enc_b_hh,
                                    WhhD_bf, dec_b_hh, Hbf);

    // 4) output projection: logits = H_all @ out_W^T + out_b  (2048 x 32000)
    {
        dim3 grid(VOUT / 256, T / 32);             // 125 x 64
        k_wmma_gemm_bf16<<<grid, 128, 0, stream>>>(Hbf, Wout_bf, out_b, out, VOUT);
    }

    // 5) in-place log_softmax per row
    k_logsoftmax<<<T, 256, 0, stream>>>(out);
}
